// QLayer_Block_55800215110165
// MI455X (gfx1250) — 8-bit matrix operand order repaired and checked by reading, compiled, not run on hardware
//
#include <hip/hip_runtime.h>
#include <hip/hip_bf16.h>

typedef int v8i __attribute__((ext_vector_type(8)));

#define Bsz 128
#define Nn  196
#define Dd  384
#define Hh  1536
#define TOK (Bsz*Nn)

#define WS_MAX   0
#define WS_SCAL  8
#define WS_W1Q   64
#define WS_B1I   448
#define WS_W2Q   832
#define WS_B2I   1216
#define WS_G1Q   1600
#define WS_G2Q   1984
#define WS_BAI   2368
#define WS_BF1I  2564
#define WS_BF2I  4100
#define WSB_PACKA  18432
#define WSB_PACKF1 71680
#define WSB_PACKF2 661504
#define WSB_X5     1251328
#define WSB_HID    20518912

#define SC_SW1 0
#define SC_SWA 1
#define SC_SG1 2
#define SC_SW2 3
#define SC_SWF1 4
#define SC_SWF2 5
#define SC_SG2 6
#define SC_M1 7
#define SC_M2 8
#define SC_M3 9
#define SC_MORG1 10
#define SC_M4 11
#define SC_M5 12
#define SC_M6 13
#define SC_M7 14
#define SC_MORG2 15
#define SC_ASOUT 16
#define SC_DEN1 17
#define SC_DENA 18
#define SC_DEN2 19
#define SC_DENF1 20
#define SC_DENF2 21

__device__ __forceinline__ float clip8s(float x)  { return fminf(fmaxf(rintf(x), -128.f), 127.f); }
__device__ __forceinline__ float clip8u(float x)  { return fminf(fmaxf(rintf(x),    0.f), 127.f); }
__device__ __forceinline__ float clip16(float x)  { return fminf(fmaxf(x, -32768.f), 32767.f); }

__device__ __forceinline__ int a_kbase(int v, int h) {
  return ((v & 3) << 2) + (h << 4) + ((v >> 2) << 5);
}

__device__ __forceinline__ v8i zero8() {
  v8i z;
#pragma unroll
  for (int e = 0; e < 8; ++e) z[e] = 0;
  return z;
}

__global__ void k_maxabs(const float* a, const float* b, const float* c,
                         const float* d, const float* e, const float* f, const float* g,
                         float* outmax) {
  const float* ptrs[7] = {a, b, c, d, e, f, g};
  const int ns[7] = {Dd, Nn * Nn, Dd, Dd, Hh * Dd, Dd * Hh, Dd};
  int which = blockIdx.x;
  const float* p = ptrs[which];
  int n = ns[which];
  __shared__ float red[256];
  float m = 0.f;
  for (int i = threadIdx.x; i < n; i += 256) m = fmaxf(m, fabsf(p[i]));
  red[threadIdx.x] = m;
  __syncthreads();
  for (int s = 128; s > 0; s >>= 1) {
    if (threadIdx.x < s) red[threadIdx.x] = fmaxf(red[threadIdx.x], red[threadIdx.x + s]);
    __syncthreads();
  }
  if (threadIdx.x == 0) outmax[which] = red[0];
}

__global__ void k_scalars(const float* a_s, const float* s_act1, const float* s_act2,
                          const float* s_add1, const float* s_act3, const float* s_mlp1,
                          const float* s_mlp2, const float* s_add2, float* wsf) {
  if (threadIdx.x != 0 || blockIdx.x != 0) return;
  const float* mx = wsf + WS_MAX;
  float* sc = wsf + WS_SCAL;
  float as = *a_s;
  float sw1 = mx[0] / 127.f, swa = mx[1] / 127.f, sg1 = mx[2] / 127.f, sw2 = mx[3] / 127.f;
  float swf1 = mx[4] / 127.f, swf2 = mx[5] / 127.f, sg2 = mx[6] / 127.f;
  sc[SC_SW1] = sw1; sc[SC_SWA] = swa; sc[SC_SG1] = sg1; sc[SC_SW2] = sw2;
  sc[SC_SWF1] = swf1; sc[SC_SWF2] = swf2; sc[SC_SG2] = sg2;
  float A1 = *s_act1, A2 = *s_act2, D1 = *s_add1, A3 = *s_act3;
  float M1 = *s_mlp1, M2 = *s_mlp2, D2 = *s_add2;
  sc[SC_M1] = (as * sw1) / A1;
  sc[SC_M2] = (A1 * swa) / A2;
  sc[SC_M3] = (A2 * sg1) / D1;  sc[SC_MORG1] = as / D1;
  sc[SC_M4] = (D1 * sw2) / A3;
  sc[SC_M5] = (A3 * swf1) / M1;
  sc[SC_M6] = (M1 * swf2) / M2;
  sc[SC_M7] = (M2 * sg2) / D2;  sc[SC_MORG2] = D1 / D2;
  sc[SC_ASOUT] = D2;
  sc[SC_DEN1] = as * sw1;  sc[SC_DENA] = A1 * swa;  sc[SC_DEN2] = D1 * sw2;
  sc[SC_DENF1] = A3 * swf1; sc[SC_DENF2] = M1 * swf2;
}

__global__ void k_chanvec(const float* n1w, const float* n1b, const float* n2w, const float* n2b,
                          const float* g1w, const float* g2w, const float* ab,
                          const float* f1b, const float* f2b, float* wsf) {
  int i = blockIdx.x * 256 + threadIdx.x;
  const float* sc = wsf + WS_SCAL;
  if (i < 384)        wsf[WS_W1Q + i]        = rintf(n1w[i] / sc[SC_SW1]);
  else if (i < 768)  { int j = i - 384;  wsf[WS_B1I + j]  = rintf(n1b[j] / sc[SC_DEN1]); }
  else if (i < 1152) { int j = i - 768;  wsf[WS_W2Q + j]  = rintf(n2w[j] / sc[SC_SW2]); }
  else if (i < 1536) { int j = i - 1152; wsf[WS_B2I + j]  = rintf(n2b[j] / sc[SC_DEN2]); }
  else if (i < 1920) { int j = i - 1536; wsf[WS_G1Q + j]  = rintf(g1w[j] / sc[SC_SG1]); }
  else if (i < 2304) { int j = i - 1920; wsf[WS_G2Q + j]  = rintf(g2w[j] / sc[SC_SG2]); }
  else if (i < 2500) { int j = i - 2304; wsf[WS_BAI + j]  = rintf(ab[j]  / sc[SC_DENA]); }
  else if (i < 4036) { int j = i - 2500; wsf[WS_BF1I + j] = rintf(f1b[j] / sc[SC_DENF1]); }
  else if (i < 4420) { int j = i - 4036; wsf[WS_BF2I + j] = rintf(f2b[j] / sc[SC_DENF2]); }
}

__global__ void k_packb(const float* __restrict__ W, const float* __restrict__ wsf, int scidx,
                        char* __restrict__ dst, int nKt, int nCol, int nK, int ld, int total) {
  int t = blockIdx.x * 256 + threadIdx.x;
  if (t >= total) return;
  float s = wsf[WS_SCAL + scidx];
  int byte = t & 3, v = (t >> 2) & 7, lane = (t >> 5) & 31, tile = t >> 10;
  int kt = tile % nKt, ct = tile / nKt;
  int col = ct * 16 + (lane & 15), hh = lane >> 4;
  int K = kt * 64 + ((v & 3) << 2) + byte + (hh << 4) + ((v >> 2) << 5);
  char q = 0;
  if (col < nCol && K < nK) q = (char)rintf(W[(size_t)col * ld + K] / s);
  dst[t] = q;
}

__global__ __launch_bounds__(128)
void k_tokenmix(const float* __restrict__ x0, const float* __restrict__ wsf,
                const char* __restrict__ packA, short* __restrict__ X5) {
  __shared__ char As[16 * 272];
  const int rt = blockIdx.x;
  const int b = rt / 24;
  const int d0 = (rt % 24) * 16;
  const float m1 = wsf[WS_SCAL + SC_M1], m2 = wsf[WS_SCAL + SC_M2];
  const float m3 = wsf[WS_SCAL + SC_M3], morg1 = wsf[WS_SCAL + SC_MORG1];
  {
    int r = threadIdx.x & 15, sub = threadIdx.x >> 4;
    int d = d0 + r;
    float w1 = wsf[WS_W1Q + d], b1 = wsf[WS_B1I + d];
    for (int n = sub; n < 256; n += 8) {
      char q = 0;
      if (n < Nn) {
        float xv = x0[((size_t)(b * Nn + n)) * Dd + d];
        q = (char)clip8s((xv * w1 + b1) * m1);
      }
      As[r * 272 + n] = q;
    }
  }
  __syncthreads();
  const int lane = threadIdx.x & 31, wid = threadIdx.x >> 5;
  const int lm = lane & 15, h = lane >> 4;
  v8i afr[4];
#pragma unroll
  for (int kt = 0; kt < 4; ++kt)
#pragma unroll
    for (int v = 0; v < 8; ++v)
      afr[kt][v] = *(const int*)&As[lm * 272 + kt * 64 + a_kbase(v, h)];
  for (int ct = wid; ct < 13; ct += 4) {
    const char* bbase = packA + (size_t)(ct * 4) * 1024 + lane * 32;
    if (ct + 4 < 13) __builtin_prefetch(bbase + 4 * 4 * 1024, 0, 1);
    v8i bfr[4];
#pragma unroll
    for (int kt = 0; kt < 4; ++kt) bfr[kt] = *(const v8i*)(bbase + kt * 1024);
    __builtin_amdgcn_sched_barrier(0);
    v8i acc = zero8();
#pragma unroll
    for (int kt = 0; kt < 4; ++kt)
      acc = __builtin_amdgcn_wmma_i32_16x16x64_iu8(true, afr[kt], true, bfr[kt], acc, false, false);
    int m = ct * 16 + lm;
    if (m < Nn) {
      float bam = wsf[WS_BAI + m];
      const float* orow = x0 + ((size_t)(b * Nn + m)) * Dd;
      short* xrow = X5 + ((size_t)(b * Nn + m)) * Dd;
#pragma unroll
      for (int v = 0; v < 8; ++v) {
        int d = d0 + v + h * 8;
        float x3 = clip8s(((float)acc[v] + bam) * m2);
        float x4 = x3 * wsf[WS_G1Q + d];
        float rsum = rintf(x4 * m3) + rintf(orow[d] * morg1);
        xrow[d] = (short)clip16(rsum);
      }
    }
  }
}

__global__ __launch_bounds__(256)
void k_fc1(const short* __restrict__ X5, const float* __restrict__ wsf,
           const char* __restrict__ packF1, char* __restrict__ HID) {
  __shared__ char As[16 * 400];
  const int tok0 = blockIdx.x * 16;
  const float m4 = wsf[WS_SCAL + SC_M4], m5 = wsf[WS_SCAL + SC_M5];
  {
    int r = threadIdx.x & 15, sub = threadIdx.x >> 4;
    const short* xr = X5 + (size_t)(tok0 + r) * Dd;
    for (int k = sub; k < Dd; k += 16) {
      float x6 = (float)xr[k] * wsf[WS_W2Q + k] + wsf[WS_B2I + k];
      As[r * 400 + k] = (char)clip8s(x6 * m4);
    }
  }
  __syncthreads();
  const int lane = threadIdx.x & 31, wid = threadIdx.x >> 5;
  const int lm = lane & 15, h = lane >> 4;
  v8i afr[6];
#pragma unroll
  for (int kt = 0; kt < 6; ++kt)
#pragma unroll
    for (int v = 0; v < 8; ++v)
      afr[kt][v] = *(const int*)&As[lm * 400 + kt * 64 + a_kbase(v, h)];
  for (int ct = wid; ct < 96; ct += 8) {
    const char* bbase = packF1 + (size_t)(ct * 6) * 1024 + lane * 32;
    if (ct + 8 < 96) __builtin_prefetch(bbase + 8 * 6 * 1024, 0, 1);
    v8i bfr[6];
#pragma unroll
    for (int kt = 0; kt < 6; ++kt) bfr[kt] = *(const v8i*)(bbase + kt * 1024);
    __builtin_amdgcn_sched_barrier(0);
    v8i acc = zero8();
#pragma unroll
    for (int kt = 0; kt < 6; ++kt)
      acc = __builtin_amdgcn_wmma_i32_16x16x64_iu8(true, afr[kt], true, bfr[kt], acc, false, false);
    int hc = ct * 16 + lm;
    float bf = wsf[WS_BF1I + hc];
#pragma unroll
    for (int v = 0; v < 8; ++v) {
      int tok = tok0 + v + h * 8;
      HID[(size_t)tok * Hh + hc] = (char)clip8u(((float)acc[v] + bf) * m5);
    }
  }
}

__global__ __launch_bounds__(256)
void k_fc2(const char* __restrict__ HID, const short* __restrict__ X5,
           const float* __restrict__ wsf, const char* __restrict__ packF2,
           float* __restrict__ out) {
  __shared__ char As[16 * 1552];
  const int tok0 = blockIdx.x * 16;
  const float m6 = wsf[WS_SCAL + SC_M6], m7 = wsf[WS_SCAL + SC_M7];
  const float morg2 = wsf[WS_SCAL + SC_MORG2];
  {
    int r = threadIdx.x & 15, sub = threadIdx.x >> 4;
    const char* hr = HID + (size_t)(tok0 + r) * Hh;
    for (int k4 = sub; k4 < 384; k4 += 16)
      *(int*)&As[r * 1552 + k4 * 4] = *(const int*)&hr[k4 * 4];
  }
  __syncthreads();
  const int lane = threadIdx.x & 31, wid = threadIdx.x >> 5;
  const int lm = lane & 15, h = lane >> 4;
  v8i acc[3];
#pragma unroll
  for (int i = 0; i < 3; ++i) acc[i] = zero8();
  for (int kc = 0; kc < 24; kc += 8) {
    v8i afr[8];
#pragma unroll
    for (int kk = 0; kk < 8; ++kk)
#pragma unroll
      for (int v = 0; v < 8; ++v)
        afr[kk][v] = *(const int*)&As[lm * 1552 + (kc + kk) * 64 + a_kbase(v, h)];
#pragma unroll
    for (int i = 0; i < 3; ++i) {
      int ct = wid + i * 8;
      const char* bbase = packF2 + (size_t)(ct * 24 + kc) * 1024 + lane * 32;
      v8i bfr[8];
#pragma unroll
      for (int kk = 0; kk < 8; ++kk) bfr[kk] = *(const v8i*)(bbase + kk * 1024);
      __builtin_amdgcn_sched_barrier(0);
#pragma unroll
      for (int kk = 0; kk < 8; ++kk)
        acc[i] = __builtin_amdgcn_wmma_i32_16x16x64_iu8(true, afr[kk], true, bfr[kk], acc[i], false, false);
    }
  }
#pragma unroll
  for (int i = 0; i < 3; ++i) {
    int ct = wid + i * 8;
    int d = ct * 16 + lm;
    float bf = wsf[WS_BF2I + d], g2 = wsf[WS_G2Q + d];
#pragma unroll
    for (int v = 0; v < 8; ++v) {
      int tok = tok0 + v + h * 8;
      float x9 = clip8s(((float)acc[i][v] + bf) * m6);
      float t2 = x9 * g2;
      float org = (float)X5[(size_t)tok * Dd + d];
      float rsum = rintf(t2 * m7) + rintf(org * morg2);
      out[(size_t)tok * Dd + d] = clip16(rsum);
    }
  }
  if (blockIdx.x == 0 && threadIdx.x == 0)
    out[(size_t)TOK * Dd] = wsf[WS_SCAL + SC_ASOUT];
}

extern "C" void kernel_launch(void* const* d_in, const int* in_sizes, int n_in,
                              void* d_out, int out_size, void* d_ws, size_t ws_size,
                              hipStream_t stream) {
  (void)in_sizes; (void)n_in; (void)out_size; (void)ws_size;
  const float* x0    = (const float*)d_in[0];
  const float* a_s   = (const float*)d_in[1];
  const float* n1w   = (const float*)d_in[2];
  const float* n1b   = (const float*)d_in[3];
  const float* aw    = (const float*)d_in[4];
  const float* ab    = (const float*)d_in[5];
  const float* g1w   = (const float*)d_in[6];
  const float* n2w   = (const float*)d_in[7];
  const float* n2b   = (const float*)d_in[8];
  const float* f1w   = (const float*)d_in[9];
  const float* f1b   = (const float*)d_in[10];
  const float* f2w   = (const float*)d_in[11];
  const float* f2b   = (const float*)d_in[12];
  const float* g2w   = (const float*)d_in[13];
  const float* sact1 = (const float*)d_in[14];
  const float* sact2 = (const float*)d_in[15];
  const float* sadd1 = (const float*)d_in[16];
  const float* sact3 = (const float*)d_in[17];
  const float* smlp1 = (const float*)d_in[18];
  const float* smlp2 = (const float*)d_in[19];
  const float* sadd2 = (const float*)d_in[20];
  float* wsf = (float*)d_ws;
  char*  wsb = (char*)d_ws;
  float* out = (float*)d_out;

  k_maxabs<<<7, 256, 0, stream>>>(n1w, aw, g1w, n2w, f1w, f2w, g2w, wsf + WS_MAX);
  k_scalars<<<1, 1, 0, stream>>>(a_s, sact1, sact2, sadd1, sact3, smlp1, smlp2, sadd2, wsf);
  k_chanvec<<<18, 256, 0, stream>>>(n1w, n1b, n2w, n2b, g1w, g2w, ab, f1b, f2b, wsf);
  k_packb<<<208, 256, 0, stream>>>(aw, wsf, SC_SWA, wsb + WSB_PACKA, 4, 196, 196, 196, 13 * 4 * 1024);
  k_packb<<<2304, 256, 0, stream>>>(f1w, wsf, SC_SWF1, wsb + WSB_PACKF1, 6, 1536, 384, 384, 96 * 6 * 1024);
  k_packb<<<2304, 256, 0, stream>>>(f2w, wsf, SC_SWF2, wsb + WSB_PACKF2, 24, 384, 1536, 1536, 24 * 24 * 1024);
  k_tokenmix<<<3072, 128, 0, stream>>>(x0, wsf, wsb + WSB_PACKA, (short*)(wsb + WSB_X5));
  k_fc1<<<1568, 256, 0, stream>>>((const short*)(wsb + WSB_X5), wsf, wsb + WSB_PACKF1, wsb + WSB_HID);
  k_fc2<<<1568, 256, 0, stream>>>((const char*)(wsb + WSB_HID), (const short*)(wsb + WSB_X5),
                                  wsf, wsb + WSB_PACKF2, out);
}
